// MultiHeadGraphAttention_13322988552558
// MI455X (gfx1250) — matrix operand order repaired and checked by reading, not compiled, not run on hardware
//
#include <hip/hip_runtime.h>

typedef __attribute__((ext_vector_type(16))) _Float16 v16h;
typedef __attribute__((ext_vector_type(8)))  _Float16 v8h;
typedef __attribute__((ext_vector_type(8)))  float    v8f;
typedef __attribute__((ext_vector_type(4)))  int      v4i;

__device__ __forceinline__ v8f wmma_acc(v16h a, v16h b, v8f c) {
  return __builtin_amdgcn_wmma_f32_16x16x32_f16(false, a, false, b, (short)0, c, false, false);
}

// A-fragment loader: row pointer + chunk ck (32 K-values) + lane group g.
// Element h holds K = h + 16*g + 32*ck  (contiguous order, matches B fills).
__device__ __forceinline__ v16h load_afrag(const _Float16* row, int ck, int g) {
  union { v16h v; v8h h[2]; } u;
  u.h[0] = *(const v8h*)(row + ck * 32 + 16 * g);
  u.h[1] = *(const v8h*)(row + ck * 32 + 16 * g + 8);
  return u.v;
}

// Cross-lane xor within a 16-lane row via DPP16 row_xmask (VALU, no LDS).
template <int M>
__device__ __forceinline__ float dpp_xor(float v) {
  int x = __builtin_amdgcn_update_dpp(0, __builtin_bit_cast(int, v),
                                      0x160 | M, 0xf, 0xf, true);
  return __builtin_bit_cast(float, x);
}
__device__ __forceinline__ float red_max16(float v) {
  v = fmaxf(v, dpp_xor<1>(v));
  v = fmaxf(v, dpp_xor<2>(v));
  v = fmaxf(v, dpp_xor<4>(v));
  v = fmaxf(v, dpp_xor<8>(v));
  return v;
}
__device__ __forceinline__ float red_sum16(float v) {
  v += dpp_xor<1>(v);
  v += dpp_xor<2>(v);
  v += dpp_xor<4>(v);
  v += dpp_xor<8>(v);
  return v;
}

// ---------------------------------------------------------------------------
// Stage 1: K/Q/V projection per head (WMMA over contraction C*KH).
// K,Q layout: [b][t][n][32] f16 (c' padded to 32, zeros via memset).
// V layout:   [b][t][32][n] f16 (pre-transposed for the attention kernel).
// ---------------------------------------------------------------------------
template <int KH>
__global__ __launch_bounds__(256) void qkv_kernel(
    const float* __restrict__ x,
    const float* __restrict__ kw, const float* __restrict__ kb,
    const float* __restrict__ qw, const float* __restrict__ qb,
    const float* __restrict__ vw, const float* __restrict__ vb,
    _Float16* __restrict__ Kf, _Float16* __restrict__ Qf, _Float16* __restrict__ Vf) {
  constexpr int CP = 25 - KH;
  constexpr int CK = 64 * KH;
  __shared__ _Float16 W[3][16 * CK];
  const int tid = threadIdx.x;
  for (int i = tid; i < 16 * CK; i += 256) {
    int t = i / CK, k = i - t * CK;
    _Float16 a = 0, b_ = 0, c_ = 0;
    if (t < 12) {
      a  = (_Float16)kw[t * CK + k];
      b_ = (_Float16)qw[t * CK + k];
      c_ = (_Float16)vw[t * CK + k];
    }
    W[0][t * CK + k] = a; W[1][t * CK + k] = b_; W[2][t * CK + k] = c_;
  }
  __syncthreads();
  const int wave = tid >> 5, lane = tid & 31, g = lane >> 4, l = lane & 15;
  const int b = blockIdx.y;
  const int pos = (blockIdx.x * 8 + wave) * 16 + l;   // pos = n*CP + c'
  const int n = pos / CP, cp = pos - n * CP;
  const long xrow = (long)b * 64 * 512 * 24;
  v8f accK = {0, 0, 0, 0, 0, 0, 0, 0};
  v8f accQ = accK, accV = accK;
#pragma unroll
  for (int ck = 0; ck < (CK >> 5); ++ck) {
    v16h bf;
#pragma unroll
    for (int h = 0; h < 16; ++h) {
      int kidx = ck * 32 + 16 * g + h;
      int ci = kidx / KH;          // compile-time divisor
      int kk = kidx - ci * KH;
      bf[h] = (_Float16)x[xrow + ((long)ci * 512 + n) * 24 + cp + kk];
    }
    accK = wmma_acc(load_afrag(&W[0][l * CK], ck, g), bf, accK);
    accQ = wmma_acc(load_afrag(&W[1][l * CK], ck, g), bf, accQ);
    accV = wmma_acc(load_afrag(&W[2][l * CK], ck, g), bf, accV);
  }
#pragma unroll
  for (int r = 0; r < 8; ++r) {
    int t = r + 8 * g;
    if (t < 12) {
      long bt = (long)b * 12 + t;
      Kf[(bt * 512 + n) * 32 + cp] = (_Float16)(accK[r] + kb[t]);
      Qf[(bt * 512 + n) * 32 + cp] = (_Float16)(accQ[r] + qb[t]);
      Vf[(bt * 32 + cp) * 512 + n] = (_Float16)(accV[r] + vb[t]);   // transposed
    }
  }
}

// ---------------------------------------------------------------------------
// Stage 2: xt[v][b][t][n][ci] = sum_ti restW_v[t][ti] * x[b][ci][n][ti]  (f16)
// v = 0..2: per-head rest_w; v = 3: mh_rest_w (temporal residual, no bias).
// ---------------------------------------------------------------------------
__global__ __launch_bounds__(256) void xt_kernel(
    const float* __restrict__ x,
    const float* __restrict__ w0, const float* __restrict__ w1,
    const float* __restrict__ w2, const float* __restrict__ w3,
    _Float16* __restrict__ xt) {
  const int v = blockIdx.y;
  const float* wp = (v == 0) ? w0 : (v == 1) ? w1 : (v == 2) ? w2 : w3;
  __shared__ float W[12 * 24];
  if (threadIdx.x < 288) W[threadIdx.x] = wp[threadIdx.x];
  __syncthreads();
  const long idx = (long)blockIdx.x * 256 + threadIdx.x;   // over B*N*C = 524288
  const int ci = (int)(idx & 63);
  const int n  = (int)((idx >> 6) & 511);
  const int b  = (int)(idx >> 15);
  const float* xr = x + (((long)b * 64 + ci) * 512 + n) * 24;
  float xl[24];
#pragma unroll
  for (int ti = 0; ti < 24; ++ti) xl[ti] = xr[ti];
#pragma unroll
  for (int t = 0; t < 12; ++t) {
    float o = 0.f;
#pragma unroll
    for (int ti = 0; ti < 24; ++ti) o += W[t * 24 + ti] * xl[ti];
    xt[((((long)v * 16 + b) * 12 + t) * 512 + n) * 64 + ci] = (_Float16)o;
  }
}

// ---------------------------------------------------------------------------
// Stage 3: flash attention per (b,t): att = softmax(K Q^T) V, all WMMA.
// V arrives pre-transposed; its 32KB tile is pulled to LDS with async-to-LDS
// B128 copies (ASYNCcnt). Grid: (192 bt, 4 row-groups); 1 row-tile per wave.
// ---------------------------------------------------------------------------
__global__ __launch_bounds__(256) void attn_kernel(
    const _Float16* __restrict__ Kf, const _Float16* __restrict__ Qf,
    const _Float16* __restrict__ Vf, _Float16* __restrict__ att) {
  __shared__ _Float16 Vt[32 * 512];                 // [c][m], straight copy
  __shared__ _Float16 Ps[8][16 * 32];
  const int tid = threadIdx.x;
  const long base = (long)blockIdx.x * (512 * 32);  // blockIdx.x = b*12+t
  {
    const _Float16* src = Vf + base;
#if __has_builtin(__builtin_amdgcn_global_load_async_to_lds_b128)
    for (int i = tid; i < 2048; i += 256) {         // 2048 x 16B chunks
      __builtin_amdgcn_global_load_async_to_lds_b128(
          (v4i*)(void*)(src + i * 8),
          (v4i*)(void*)(Vt + i * 8),
          0, 0);
    }
#if __has_builtin(__builtin_amdgcn_s_wait_asynccnt)
    __builtin_amdgcn_s_wait_asynccnt(0);
#else
    asm volatile("s_wait_asynccnt 0" ::: "memory");
#endif
#else
    for (int i = tid; i < 2048; i += 256)
      *(uint4*)(Vt + i * 8) = *(const uint4*)(src + i * 8);
#endif
  }
  __syncthreads();
  const int wave = tid >> 5, lane = tid & 31, g = lane >> 4, l = lane & 15;
  _Float16* P = &Ps[wave][0];
  const int n0 = 16 * (blockIdx.y * 8 + wave);
  v16h ak = load_afrag(Kf + base + (long)(n0 + l) * 32, 0, g);
  v8f o0 = {0, 0, 0, 0, 0, 0, 0, 0}, o1 = o0;
  float rmax[8], rsum[8];
#pragma unroll
  for (int r = 0; r < 8; ++r) { rmax[r] = -3.0e38f; rsum[r] = 0.0f; }
  for (int m0 = 0; m0 < 512; m0 += 32) {
    v16h bq0 = *(const v16h*)(Qf + base + (long)(m0 + l) * 32 + 16 * g);
    v16h bq1 = *(const v16h*)(Qf + base + (long)(m0 + 16 + l) * 32 + 16 * g);
    v8f zero = {0, 0, 0, 0, 0, 0, 0, 0};
    v8f s0 = wmma_acc(ak, bq0, zero);
    v8f s1 = wmma_acc(ak, bq1, zero);
#pragma unroll
    for (int r = 0; r < 8; ++r) {
      float mx = red_max16(fmaxf(s0[r], s1[r]));
      float nm = fmaxf(rmax[r], mx);
      float sc = __expf(rmax[r] - nm);
      float p0 = __expf(s0[r] - nm);
      float p1 = __expf(s1[r] - nm);
      float ts = red_sum16(p0 + p1);
      rsum[r] = rsum[r] * sc + ts;
      rmax[r] = nm;
      o0[r] *= sc; o1[r] *= sc;
      int row = r + 8 * g;
      P[row * 32 + l] = (_Float16)p0;
      P[row * 32 + 16 + l] = (_Float16)p1;
    }
    asm volatile("s_wait_dscnt 0" ::: "memory");
    v16h ap = load_afrag(&P[l * 32], 0, g);
    v16h bv0 = *(const v16h*)(&Vt[(long)l * 512 + m0 + 16 * g]);
    v16h bv1 = *(const v16h*)(&Vt[(long)(16 + l) * 512 + m0 + 16 * g]);
    o0 = wmma_acc(ap, bv0, o0);
    o1 = wmma_acc(ap, bv1, o1);
  }
#pragma unroll
  for (int r = 0; r < 8; ++r) {
    float inv = 1.0f / rsum[r];
    long n = n0 + r + 8 * g;
    att[base + n * 32 + l] = (_Float16)(o0[r] * inv);
    att[base + n * 32 + 16 + l] = (_Float16)(o1[r] * inv);
  }
}

// ---------------------------------------------------------------------------
// Stage 4: per-head fused epilogue (pre-BN):
//   Y[pos][head*64+e] = fcoutW[e,:CP].att + resW[e,:].xt
//                       + fcout_b[e] + res_b[e]*sum(rest_w[t,:]) + rest_b[t]
// Contraction 96 = 3 WMMA chunks (32 att-pad + 64 xt).
// ---------------------------------------------------------------------------
__global__ __launch_bounds__(256) void head_out_kernel(
    const _Float16* __restrict__ att, const _Float16* __restrict__ xt,
    const float* __restrict__ fcw, const float* __restrict__ fcb,
    const float* __restrict__ rw, const float* __restrict__ rb,
    const float* __restrict__ restw, const float* __restrict__ restb,
    float* __restrict__ Y, int head, int CP) {
  __shared__ _Float16 Wc[64 * 96];
  __shared__ float srw[12];
  const int tid = threadIdx.x;
  for (int i = tid; i < 64 * 96; i += 256) {
    int e = i / 96, k = i - e * 96;
    float w = (k < 32) ? ((k < CP) ? fcw[e * CP + k] : 0.0f) : rw[e * 64 + (k - 32)];
    Wc[i] = (_Float16)w;
  }
  if (tid < 12) {
    float s = 0.f;
    for (int ti = 0; ti < 24; ++ti) s += restw[tid * 24 + ti];
    srw[tid] = s;
  }
  __syncthreads();
  const int wave = tid >> 5, lane = tid & 31, g = lane >> 4, l = lane & 15;
  const long pos = ((long)blockIdx.x * 8 + wave) * 16 + l;   // b*6144 + n*12 + t
  const int b = (int)(pos / 6144);
  const int rem = (int)(pos - (long)b * 6144);
  const int n = rem / 12, t = rem - n * 12;
  const long ab = (((long)b * 12 + t) * 512 + n) * 32;
  const long xb = (((long)b * 12 + t) * 512 + n) * 64;
  v16h b0 = *(const v16h*)(att + ab + 16 * g);
  v16h b1 = *(const v16h*)(xt + xb + 16 * g);
  v16h b2 = *(const v16h*)(xt + xb + 32 + 16 * g);
#pragma unroll
  for (int mt = 0; mt < 4; ++mt) {
    const _Float16* wr = &Wc[(mt * 16 + l) * 96];
    v8f acc = {0, 0, 0, 0, 0, 0, 0, 0};
    acc = wmma_acc(load_afrag(wr, 0, g), b0, acc);
    acc = wmma_acc(load_afrag(wr, 1, g), b1, acc);
    acc = wmma_acc(load_afrag(wr, 2, g), b2, acc);
#pragma unroll
    for (int r = 0; r < 8; ++r) {
      int e = mt * 16 + r + 8 * g;
      Y[pos * 192 + head * 64 + e] = acc[r] + fcb[e] + rb[e] * srw[t] + restb[t];
    }
  }
}

// ---------------------------------------------------------------------------
// Batch-norm statistics (biased var, matches jnp.var): one block per channel.
// addr(ch, i) = ch*cs + (i/ic)*os + (i%ic)*is
// ---------------------------------------------------------------------------
__global__ __launch_bounds__(256) void bn_stats_kernel(
    const float* __restrict__ src, long cs, long ic, long os, long is,
    long count, float* __restrict__ mean, float* __restrict__ rstd) {
  __shared__ float s1s[256], s2s[256];
  const int tid = threadIdx.x;
  const long base = (long)blockIdx.x * cs;
  float s1 = 0.f, s2 = 0.f;
  for (long i = tid; i < count; i += 256) {
    long q = i / ic;
    float v = src[base + q * os + (i - q * ic) * is];
    s1 += v; s2 += v * v;
  }
  s1s[tid] = s1; s2s[tid] = s2;
  __syncthreads();
  for (int off = 128; off > 0; off >>= 1) {
    if (tid < off) { s1s[tid] += s1s[tid + off]; s2s[tid] += s2s[tid + off]; }
    __syncthreads();
  }
  if (tid == 0) {
    float m = s1s[0] / (float)count;
    float var = s2s[0] / (float)count - m * m;
    mean[blockIdx.x] = m;
    rstd[blockIdx.x] = rsqrtf(var + 1e-5f);
  }
}

// ---------------------------------------------------------------------------
// Stage 5: final projection. Normalize Y per-channel (head BN) while building
// B-fragments, WMMA over 192 channels, add mh bias + temporal residual.
// Z laid out [b][e][n][t] (= output layout, pre final BN).
// ---------------------------------------------------------------------------
__global__ __launch_bounds__(256) void final_kernel(
    const float* __restrict__ Y, const float* __restrict__ meanY, const float* __restrict__ rstdY,
    const float* __restrict__ g0, const float* __restrict__ be0,
    const float* __restrict__ g1, const float* __restrict__ be1,
    const float* __restrict__ g2, const float* __restrict__ be2,
    const float* __restrict__ mhw, const float* __restrict__ mhb,
    const float* __restrict__ mrestb, const _Float16* __restrict__ xt3,
    float* __restrict__ Z) {
  __shared__ _Float16 Wc[64 * 192];
  __shared__ float gam[192], bet[192], mu[192], rs[192];
  const int tid = threadIdx.x;
  for (int i = tid; i < 64 * 192; i += 256) Wc[i] = (_Float16)mhw[i];
  if (tid < 192) {
    int hd = tid >> 6, e = tid & 63;
    const float* gp = (hd == 0) ? g0 : ((hd == 1) ? g1 : g2);
    const float* bp = (hd == 0) ? be0 : ((hd == 1) ? be1 : be2);
    gam[tid] = gp[e]; bet[tid] = bp[e]; mu[tid] = meanY[tid]; rs[tid] = rstdY[tid];
  }
  __syncthreads();
  const int wave = tid >> 5, lane = tid & 31, g = lane >> 4, l = lane & 15;
  const long pos = ((long)blockIdx.x * 8 + wave) * 16 + l;
  const int b = (int)(pos / 6144);
  const int rem = (int)(pos - (long)b * 6144);
  const int n = rem / 12, t = rem - n * 12;
  v16h bf[6];
#pragma unroll
  for (int ck = 0; ck < 6; ++ck) {
#pragma unroll
    for (int h = 0; h < 16; ++h) {
      int he = ck * 32 + 16 * g + h;
      float yv = Y[pos * 192 + he];
      bf[ck][h] = (_Float16)((yv - mu[he]) * rs[he] * gam[he] + bet[he]);
    }
  }
  const long xb = (((long)b * 12 + t) * 512 + n) * 64;
#pragma unroll
  for (int mt = 0; mt < 4; ++mt) {
    const _Float16* wr = &Wc[(mt * 16 + l) * 192];
    v8f acc = {0, 0, 0, 0, 0, 0, 0, 0};
#pragma unroll
    for (int ck = 0; ck < 6; ++ck) acc = wmma_acc(load_afrag(wr, ck, g), bf[ck], acc);
#pragma unroll
    for (int r = 0; r < 8; ++r) {
      int e = mt * 16 + r + 8 * g;
      float v = acc[r] + mhb[e] + mrestb[t] + (float)xt3[xb + e];
      Z[(((long)b * 64 + e) * 512 + n) * 12 + t] = v;
    }
  }
}

// Stage 6: apply final BN -> d_out.
__global__ __launch_bounds__(256) void norm_kernel(
    const float* __restrict__ Z, const float* __restrict__ mean, const float* __restrict__ rstd,
    const float* __restrict__ gg, const float* __restrict__ bb, float* __restrict__ out) {
  int i = blockIdx.x * 256 + threadIdx.x;        // 6291456 total
  int e = (i / 6144) & 63;
  out[i] = (Z[i] - mean[e]) * rstd[e] * gg[e] + bb[e];
}

// ---------------------------------------------------------------------------
extern "C" void kernel_launch(void* const* d_in, const int* in_sizes, int n_in,
                              void* d_out, int out_size, void* d_ws, size_t ws_size,
                              hipStream_t stream) {
  (void)in_sizes; (void)n_in; (void)out_size; (void)ws_size;
  const float* X = (const float*)d_in[0];
  auto H = [&](int i, int j) { return (const float*)d_in[1 + 14 * i + j]; };
  // per-head param slots: 0 kw 1 kb 2 qw 3 qb 4 vw 5 vb 6 fcout_w 7 fcout_b
  //                       8 res_w 9 res_b 10 rest_w 11 rest_b 12 bn_g 13 bn_b
  const float* mh_rest_w = (const float*)d_in[43];
  const float* mh_rest_b = (const float*)d_in[44];
  const float* mh_fcw    = (const float*)d_in[45];
  const float* mh_fcb    = (const float*)d_in[46];
  const float* mh_bn_g   = (const float*)d_in[47];
  const float* mh_bn_b   = (const float*)d_in[48];

  char* ws = (char*)d_ws;
  const long HB = 6291456L;    // bytes of one [B,T,N,32] f16 tensor
  const long XB = 12582912L;   // bytes of one [B,T,N,64] f16 tensor
  _Float16 *Kf[3], *Qf[3], *Vf[3], *At[3];
  for (int i = 0; i < 3; ++i) {
    Kf[i] = (_Float16*)(ws + (long)i * HB);
    Qf[i] = (_Float16*)(ws + (long)(3 + i) * HB);
    Vf[i] = (_Float16*)(ws + (long)(6 + i) * HB);
    At[i] = (_Float16*)(ws + (long)(9 + i) * HB);
  }
  _Float16* XT = (_Float16*)(ws + 12L * HB);
  float* Y = (float*)(ws + 12L * HB + 4L * XB);          // [98304][192] f32
  float* Z = (float*)((char*)Y + 75497472L);             // [16][64][512][12] f32
  float* meanY = (float*)((char*)Z + 25165824L);
  float* rstdY = meanY + 192;
  float* meanZ = rstdY + 192;
  float* rstdZ = meanZ + 64;

  // zero K/Q/V/att buffers (c' padding beyond Cp must be 0)
  (void)hipMemsetAsync(ws, 0, 12L * HB, stream);

  // QKV projections (templated on kernel width)
  {
    int tiles;
    tiles = (512 * 23) / 16;
    qkv_kernel<2><<<dim3(tiles / 8, 16), 256, 0, stream>>>(
        X, H(0, 0), H(0, 1), H(0, 2), H(0, 3), H(0, 4), H(0, 5), Kf[0], Qf[0], Vf[0]);
    tiles = (512 * 22) / 16;
    qkv_kernel<3><<<dim3(tiles / 8, 16), 256, 0, stream>>>(
        X, H(1, 0), H(1, 1), H(1, 2), H(1, 3), H(1, 4), H(1, 5), Kf[1], Qf[1], Vf[1]);
    tiles = (512 * 19) / 16;
    qkv_kernel<6><<<dim3(tiles / 8, 16), 256, 0, stream>>>(
        X, H(2, 0), H(2, 1), H(2, 2), H(2, 3), H(2, 4), H(2, 5), Kf[2], Qf[2], Vf[2]);
  }

  xt_kernel<<<dim3(2048, 4), 256, 0, stream>>>(X, H(0, 10), H(1, 10), H(2, 10),
                                               mh_rest_w, XT);

  for (int i = 0; i < 3; ++i)
    attn_kernel<<<dim3(192, 4), 256, 0, stream>>>(Kf[i], Qf[i], Vf[i], At[i]);

  const long XTH = XB / 2;   // halves per xt variant
  const int KS[3] = {2, 3, 6};
  for (int i = 0; i < 3; ++i) {
    int CP = 24 - KS[i] + 1;
    head_out_kernel<<<768, 256, 0, stream>>>(
        At[i], XT + (long)i * XTH,
        H(i, 6), H(i, 7), H(i, 8), H(i, 9), H(i, 10), H(i, 11),
        Y, i, CP);
  }

  // head BN stats: 192 channels over 98304 positions; addr = ch + i*192
  bn_stats_kernel<<<192, 256, 0, stream>>>(Y, 1L, 98304L, 0L, 192L, 98304L,
                                           meanY, rstdY);

  final_kernel<<<768, 256, 0, stream>>>(
      Y, meanY, rstdY,
      H(0, 12), H(0, 13), H(1, 12), H(1, 13), H(2, 12), H(2, 13),
      mh_fcw, mh_fcb, mh_rest_b, XT + 3L * XTH, Z);

  // final BN stats: 64 channels; addr = e*6144 + (i/6144)*393216 + i%6144
  bn_stats_kernel<<<64, 256, 0, stream>>>(Z, 6144L, 6144L, 393216L, 1L, 98304L,
                                          meanZ, rstdZ);

  norm_kernel<<<24576, 256, 0, stream>>>(Z, meanZ, rstdZ, mh_bn_g, mh_bn_b,
                                         (float*)d_out);
}
